// GenomicGNN_15255723836181
// MI455X (gfx1250) — compile-verified, matrix operand order checked by reading, not run on hardware
//
#include <hip/hip_runtime.h>
#include <hip/hip_bf16.h>

// ---------------------------------------------------------------------------
// GenomicGNN pipeline for MI455X (gfx1250, wave32).
// GEMMs use v_wmma_f32_16x16x32_bf16 (bf16 in, fp32 accumulate) with
// fragment-prepacked B operands (pure b128 loads) and LDS-staged coalesced
// stores. Graph aggregation uses f32 global atomics; self-loops folded
// analytically into the finalize kernels.
// ---------------------------------------------------------------------------

typedef __attribute__((ext_vector_type(16))) __bf16 v16bf;
typedef __attribute__((ext_vector_type(8)))  float  v8f;

#define H_DIM   64
#define HEADS   4
#define GH      (HEADS * H_DIM)   // 256
#define NGRAPH  64

// ------------------------------- utilities --------------------------------

__global__ void gg_fill_f32(float* __restrict__ p, float v, long long n) {
  long long t = (long long)blockIdx.x * blockDim.x + threadIdx.x;
  if (t < n) p[t] = v;
}

__global__ void gg_f32_to_bf16(const float* __restrict__ s, __bf16* __restrict__ d,
                               long long n) {
  long long t = (long long)blockIdx.x * blockDim.x + threadIdx.x;
  if (t < n) d[t] = (__bf16)s[t];
}

__device__ __forceinline__ float gg_leaky(float v) { return v > 0.f ? v : 0.2f * v; }

__device__ __forceinline__ void gg_atomic_max_f32(float* addr, float val) {
  // IEEE ordering trick: int-max for non-negative, uint-min for negative.
  if (val >= 0.f) atomicMax((int*)addr, __float_as_int(val));
  else            atomicMin((unsigned int*)addr, __float_as_uint(val));
}

// edge id -> (src,dst); ids >= E are the implicit self-loops
__device__ __forceinline__ void gg_edge(const int* __restrict__ ei, long long E,
                                        long long eid, int& s, int& d) {
  if (eid < E) { s = ei[eid]; d = ei[E + eid]; }
  else         { s = d = (int)(eid - E); }
}

// --------------------- B-operand fragment pre-packing ----------------------
// Pack fp32 B[K,NC] into bf16 WMMA-B fragment order:
//   Bp[((ntile*(K/32) + kstep)*32 + lane)*16 + elem]
// lane: n = ntile*16 + (lane&15), ko = (lane>>4)*8
// elem i<8  -> k = kstep*32 + ko + i
// elem i>=8 -> k = kstep*32 + ko + 16 + (i-8)
__global__ void gg_pack_b(const float* __restrict__ B, __bf16* __restrict__ Bp,
                          int K, int NC) {
  long long t = (long long)blockIdx.x * blockDim.x + threadIdx.x;
  if (t >= (long long)K * NC) return;
  const int elem = (int)(t & 15);
  const int lane = (int)((t >> 4) & 31);
  long long rest = t >> 9;
  const int ksteps = K >> 5;
  const int kstep = (int)(rest % ksteps);
  const int ntile = (int)(rest / ksteps);
  const int n  = ntile * 16 + (lane & 15);
  const int ko = (lane >> 4) * 8;
  const int k  = kstep * 32 + ko + (elem < 8 ? elem : 16 + (elem - 8));
  Bp[t] = (__bf16)B[(size_t)k * NC + n];
}

// ------------------------- WMMA bf16 GEMM (fp32 acc) -----------------------
// C[M,NC] = A[M,K] * Bpacked; K % 32 == 0, NC % 64 == 0.
// grid = (ceil(M/16), NC/64), block = 128 (4 waves; wave w -> 16-col tile w).
// Inner loop: 2x b128 loads for A, 2x b128 loads for packed B, one WMMA.
// Epilogue: stage 16x64 tile in LDS, coalesced float4-pair writeback.
__global__ __launch_bounds__(128)
void gg_gemm_bf16(const __bf16* __restrict__ A, const __bf16* __restrict__ Bp,
                  float* __restrict__ C, int M, int K, int NC) {
  __shared__ float lds[16 * 64];
  const int lane = threadIdx.x & 31;
  const int wave = threadIdx.x >> 5;
  const int mbase = blockIdx.x * 16;
  const int ntile = blockIdx.y * 4 + wave;     // global 16-col tile id
  const int lo15 = lane & 15;
  const int hi   = lane >> 4;                  // 0 | 1
  const int ko   = hi * 8;

  int m = mbase + lo15;
  if (m >= M) m = M - 1;                       // load clamp only
  const int ksteps = K >> 5;
  const __bf16* __restrict__ bp =
      Bp + ((size_t)ntile * ksteps * 32 + lane) * 16;

  v8f acc = {};
  for (int ks = 0; ks < ksteps; ++ks) {
    v16bf a, b;
    const __bf16* rowA = A + (size_t)m * K + ks * 32 + ko;
    ((uint4*)&a)[0] = *(const uint4*)(rowA);
    ((uint4*)&a)[1] = *(const uint4*)(rowA + 16);
    const __bf16* pb = bp + (size_t)ks * 32 * 16;
    ((uint4*)&b)[0] = ((const uint4*)pb)[0];
    ((uint4*)&b)[1] = ((const uint4*)pb)[1];
    acc = __builtin_amdgcn_wmma_f32_16x16x32_bf16(
        /*neg_a=*/false, a, /*neg_b=*/false, b,
        /*c_mod=*/(short)0, acc, /*reuse_a=*/false, /*reuse_b=*/false);
  }

  // stage to LDS: lane owns column (wave*16 + lo15); VGPR r -> row r + 8*hi
  const int c = wave * 16 + lo15;
#pragma unroll
  for (int r = 0; r < 8; ++r) lds[(r + 8 * hi) * 64 + c] = acc[r];
  __syncthreads();

  // coalesced writeback: thread t -> (row = t/8, 8 consecutive floats)
  const int row = threadIdx.x >> 3;
  const int c8  = (threadIdx.x & 7) * 8;
  if (mbase + row < M) {
    float* dst = C + (size_t)(mbase + row) * NC + blockIdx.y * 64 + c8;
    const float4* src = (const float4*)&lds[row * 64 + c8];
    ((float4*)dst)[0] = src[0];
    ((float4*)dst)[1] = src[1];
  }
}

// ------------------------------ GCN kernels --------------------------------

__global__ void gg_degree(const int* __restrict__ ei, float* __restrict__ deg, int E) {
  int e = blockIdx.x * blockDim.x + threadIdx.x;
  if (e < E) atomicAdd(&deg[ei[E + e]], 1.0f);
}

__global__ void gg_dinv(float* __restrict__ deg, int N) {
  int i = blockIdx.x * blockDim.x + threadIdx.x;
  if (i < N) deg[i] = rsqrtf(deg[i] + 1.0f);   // +1 self-loop; deg>=1 always
}

// one thread = (edge, 4 features); agg[dst] += y[src]*dinv[src]*dinv[dst]
__global__ void gg_gcn_scatter(const float* __restrict__ y, const int* __restrict__ ei,
                               const float* __restrict__ dinv, float* __restrict__ agg,
                               int E) {
  long long t = (long long)blockIdx.x * blockDim.x + threadIdx.x;
  long long e = t >> 4;
  if (e >= E) return;
  const int f4 = (int)(t & 15) * 4;
  const int s = ei[e], d = ei[E + e];
  const float nrm = dinv[s] * dinv[d];
  const float4 v = *(const float4*)(y + (size_t)s * H_DIM + f4);
  float* o = agg + (size_t)d * H_DIM + f4;
  atomicAdd(o + 0, v.x * nrm);
  atomicAdd(o + 1, v.y * nrm);
  atomicAdd(o + 2, v.z * nrm);
  atomicAdd(o + 3, v.w * nrm);
}

// agg = relu(agg + y*dinv^2 + b)   (self-loop contribution folded in)
__global__ void gg_gcn_finalize(float* __restrict__ agg, const float* __restrict__ y,
                                const float* __restrict__ dinv,
                                const float* __restrict__ b, int N) {
  long long t = (long long)blockIdx.x * blockDim.x + threadIdx.x;
  if (t >= (long long)N * H_DIM) return;
  const int i = (int)(t >> 6), f = (int)(t & 63);
  const float di = dinv[i];
  agg[t] = fmaxf(agg[t] + y[t] * di * di + b[f], 0.0f);
}

// ------------------------------ GAT kernels --------------------------------

__global__ void gg_att_scores(const float* __restrict__ hh,
                              const float* __restrict__ att_src,
                              const float* __restrict__ att_dst,
                              float* __restrict__ a_s, float* __restrict__ a_d, int N) {
  long long t = (long long)blockIdx.x * blockDim.x + threadIdx.x;
  if (t >= (long long)N * HEADS) return;
  const int nI = (int)(t >> 2), h = (int)(t & 3);
  const float* v  = hh + (size_t)nI * GH + h * H_DIM;
  const float* as = att_src + h * H_DIM;
  const float* ad = att_dst + h * H_DIM;
  float ss = 0.f, sd = 0.f;
#pragma unroll 8
  for (int c = 0; c < H_DIM; ++c) { const float x = v[c]; ss += x * as[c]; sd += x * ad[c]; }
  a_s[t] = ss; a_d[t] = sd;
}

__global__ void gg_edge_max(const int* __restrict__ ei, const float* __restrict__ a_s,
                            const float* __restrict__ a_d, float* __restrict__ maxv,
                            long long E, long long Etot) {
  long long t = (long long)blockIdx.x * blockDim.x + threadIdx.x;
  long long e = t >> 2;
  if (e >= Etot) return;
  const int h = (int)(t & 3);
  int s, d; gg_edge(ei, E, e, s, d);
  gg_atomic_max_f32(&maxv[(size_t)d * HEADS + h],
                    gg_leaky(a_s[(size_t)s * HEADS + h] + a_d[(size_t)d * HEADS + h]));
}

__global__ void gg_edge_exp(const int* __restrict__ ei, const float* __restrict__ a_s,
                            const float* __restrict__ a_d, const float* __restrict__ maxv,
                            float* __restrict__ alpha, float* __restrict__ denom,
                            long long E, long long Etot) {
  long long t = (long long)blockIdx.x * blockDim.x + threadIdx.x;
  long long e = t >> 2;
  if (e >= Etot) return;
  const int h = (int)(t & 3);
  int s, d; gg_edge(ei, E, e, s, d);
  const float v = gg_leaky(a_s[(size_t)s * HEADS + h] + a_d[(size_t)d * HEADS + h]);
  const float w = expf(v - maxv[(size_t)d * HEADS + h]);
  alpha[e * HEADS + h] = w;
  atomicAdd(&denom[(size_t)d * HEADS + h], w);
}

__global__ void gg_alpha_norm(const int* __restrict__ ei, float* __restrict__ alpha,
                              const float* __restrict__ denom, long long E, long long Etot) {
  long long t = (long long)blockIdx.x * blockDim.x + threadIdx.x;
  long long e = t >> 2;
  if (e >= Etot) return;
  const int h = (int)(t & 3);
  int s, d; gg_edge(ei, E, e, s, d);
  alpha[e * HEADS + h] /= denom[(size_t)d * HEADS + h];
}

// one thread = (edge, feature); loops over the 4 heads
__global__ void gg_gat_aggr(const int* __restrict__ ei, const float* __restrict__ hh,
                            const float* __restrict__ alpha, float* __restrict__ gat,
                            long long E, long long Etot) {
  long long t = (long long)blockIdx.x * blockDim.x + threadIdx.x;
  long long e = t >> 6;
  if (e >= Etot) return;
  const int f = (int)(t & 63);
  int s, d; gg_edge(ei, E, e, s, d);
#pragma unroll
  for (int h = 0; h < HEADS; ++h) {
    const float a = alpha[e * HEADS + h];
    atomicAdd(&gat[(size_t)d * GH + h * H_DIM + f],
              hh[(size_t)s * GH + h * H_DIM + f] * a);
  }
}

// ------------------------------ pool + MLP ---------------------------------

__global__ void gg_pool(const float* __restrict__ gat, const float* __restrict__ bg,
                        const int* __restrict__ batch, float* __restrict__ pool,
                        float* __restrict__ cnt, int N) {
  long long t = (long long)blockIdx.x * blockDim.x + threadIdx.x;
  if (t >= (long long)N * GH) return;
  const int nI = (int)(t >> 8), c = (int)(t & 255);
  const int g = batch[nI];
  atomicAdd(&pool[(size_t)g * GH + c], gat[t] + bg[c]);
  if (c == 0) atomicAdd(&cnt[g], 1.0f);
}

__global__ void gg_mlp(const float* __restrict__ pool, const float* __restrict__ cnt,
                       const float* __restrict__ Wf1, const float* __restrict__ bf1,
                       const float* __restrict__ Wf2, const float* __restrict__ bf2,
                       float* __restrict__ out) {
  const int g = threadIdx.x;
  if (g >= NGRAPH) return;
  const float inv = 1.0f / fmaxf(cnt[g], 1.0f);
  const float* p = pool + (size_t)g * GH;
  float acc = 0.f;
  for (int j = 0; j < H_DIM; ++j) {
    float s = 0.f;
    for (int k = 0; k < GH; ++k) s += p[k] * Wf1[(size_t)k * H_DIM + j];
    s = fmaxf(s * inv + bf1[j], 0.0f);
    acc += s * Wf2[j];
  }
  out[g] = acc + bf2[0];
}

// ------------------------------- launcher ----------------------------------

static inline int gg_nblk(long long n, int b) { return (int)((n + b - 1) / b); }

extern "C" void kernel_launch(void* const* d_in, const int* in_sizes, int n_in,
                              void* d_out, int out_size, void* d_ws, size_t ws_size,
                              hipStream_t stream) {
  const float* x       = (const float*)d_in[0];
  const int*   ei      = (const int*)  d_in[1];
  const int*   batch   = (const int*)  d_in[2];
  const float* W1      = (const float*)d_in[3];
  const float* b1      = (const float*)d_in[4];
  const float* W2      = (const float*)d_in[5];
  const float* b2      = (const float*)d_in[6];
  const float* Wg      = (const float*)d_in[7];
  const float* att_src = (const float*)d_in[8];
  const float* att_dst = (const float*)d_in[9];
  const float* bg      = (const float*)d_in[10];
  const float* Wf1     = (const float*)d_in[11];
  const float* bf1     = (const float*)d_in[12];
  const float* Wf2     = (const float*)d_in[13];
  const float* bf2     = (const float*)d_in[14];
  float* out = (float*)d_out;

  const int N = in_sizes[2];            // 50000 (multiple of 16)
  const int E = in_sizes[1] / 2;        // 800000
  const int F = in_sizes[0] / N;        // 256
  const long long Etot = (long long)E + N;   // with self-loops

  // ---- workspace layout (~164 MB; alpha aliases the dead x_bf16 region) ----
  char* ws = (char*)d_ws;
  size_t off = 0;
  auto carve = [&](size_t bytes) { size_t o = off; off += (bytes + 255) & ~(size_t)255; return o; };
  float*  dinv = (float*) (ws + carve((size_t)N * 4));
  __bf16* xb   = (__bf16*)(ws + carve((size_t)N * F * 2));       // reused as alpha later
  __bf16* w1p  = (__bf16*)(ws + carve((size_t)F * H_DIM * 2));   // packed B fragments
  __bf16* w2p  = (__bf16*)(ws + carve((size_t)H_DIM * H_DIM * 2));
  __bf16* wgp  = (__bf16*)(ws + carve((size_t)H_DIM * GH * 2));
  float*  y    = (float*) (ws + carve((size_t)N * H_DIM * 4));
  float*  agg  = (float*) (ws + carve((size_t)N * H_DIM * 4));   // becomes h1/h2
  __bf16* hb   = (__bf16*)(ws + carve((size_t)N * H_DIM * 2));
  float*  hh   = (float*) (ws + carve((size_t)N * GH * 4));
  float*  a_s  = (float*) (ws + carve((size_t)N * HEADS * 4));
  float*  a_d  = (float*) (ws + carve((size_t)N * HEADS * 4));
  float*  maxv = (float*) (ws + carve((size_t)N * HEADS * 4));
  float*  den  = (float*) (ws + carve((size_t)N * HEADS * 4));
  float*  gat  = (float*) (ws + carve((size_t)N * GH * 4));
  float*  pool = (float*) (ws + carve((size_t)NGRAPH * GH * 4));
  float*  cnt  = (float*) (ws + carve((size_t)NGRAPH * 4));
  float*  alpha = (float*)xb;   // (E+N)*4 floats = 13.6MB <= 25.6MB region
  (void)ws_size; (void)n_in; (void)out_size;

  const int B = 256;

  // ---- degrees / symmetric norm ----
  gg_fill_f32<<<gg_nblk(N, B), B, 0, stream>>>(dinv, 0.f, N);
  gg_degree<<<gg_nblk(E, B), B, 0, stream>>>(ei, dinv, E);
  gg_dinv<<<gg_nblk(N, B), B, 0, stream>>>(dinv, N);

  // ---- bf16 conversion of x; weights converted+packed to fragment order ----
  gg_f32_to_bf16<<<gg_nblk((long long)N * F, B), B, 0, stream>>>(x, xb, (long long)N * F);
  gg_pack_b<<<gg_nblk((long long)F * H_DIM, B), B, 0, stream>>>(W1, w1p, F, H_DIM);
  gg_pack_b<<<gg_nblk((long long)H_DIM * H_DIM, B), B, 0, stream>>>(W2, w2p, H_DIM, H_DIM);
  gg_pack_b<<<gg_nblk((long long)H_DIM * GH, B), B, 0, stream>>>(Wg, wgp, H_DIM, GH);

  // ---- GCN layer 1: y = x@W1 (WMMA), scatter, finalize ----
  gg_gemm_bf16<<<dim3((N + 15) / 16, 1), 128, 0, stream>>>(xb, w1p, y, N, F, H_DIM);
  gg_fill_f32<<<gg_nblk((long long)N * H_DIM, B), B, 0, stream>>>(agg, 0.f, (long long)N * H_DIM);
  gg_gcn_scatter<<<gg_nblk((long long)E * 16, B), B, 0, stream>>>(y, ei, dinv, agg, E);
  gg_gcn_finalize<<<gg_nblk((long long)N * H_DIM, B), B, 0, stream>>>(agg, y, dinv, b1, N);

  // ---- GCN layer 2 ----
  gg_f32_to_bf16<<<gg_nblk((long long)N * H_DIM, B), B, 0, stream>>>(agg, hb, (long long)N * H_DIM);
  gg_gemm_bf16<<<dim3((N + 15) / 16, 1), 128, 0, stream>>>(hb, w2p, y, N, H_DIM, H_DIM);
  gg_fill_f32<<<gg_nblk((long long)N * H_DIM, B), B, 0, stream>>>(agg, 0.f, (long long)N * H_DIM);
  gg_gcn_scatter<<<gg_nblk((long long)E * 16, B), B, 0, stream>>>(y, ei, dinv, agg, E);
  gg_gcn_finalize<<<gg_nblk((long long)N * H_DIM, B), B, 0, stream>>>(agg, y, dinv, b2, N);

  // ---- GAT: hh = h@Wg (WMMA), attention softmax per dst, aggregate ----
  gg_f32_to_bf16<<<gg_nblk((long long)N * H_DIM, B), B, 0, stream>>>(agg, hb, (long long)N * H_DIM);
  gg_gemm_bf16<<<dim3((N + 15) / 16, GH / 64), 128, 0, stream>>>(hb, wgp, hh, N, H_DIM, GH);
  gg_att_scores<<<gg_nblk((long long)N * HEADS, B), B, 0, stream>>>(hh, att_src, att_dst, a_s, a_d, N);
  gg_fill_f32<<<gg_nblk((long long)N * HEADS, B), B, 0, stream>>>(maxv, -1e30f, (long long)N * HEADS);
  gg_fill_f32<<<gg_nblk((long long)N * HEADS, B), B, 0, stream>>>(den, 0.f, (long long)N * HEADS);
  gg_edge_max<<<gg_nblk(Etot * HEADS, B), B, 0, stream>>>(ei, a_s, a_d, maxv, E, Etot);
  gg_edge_exp<<<gg_nblk(Etot * HEADS, B), B, 0, stream>>>(ei, a_s, a_d, maxv, alpha, den, E, Etot);
  gg_alpha_norm<<<gg_nblk(Etot * HEADS, B), B, 0, stream>>>(ei, alpha, den, E, Etot);
  gg_fill_f32<<<gg_nblk((long long)N * GH, B), B, 0, stream>>>(gat, 0.f, (long long)N * GH);
  gg_gat_aggr<<<gg_nblk(Etot * 64, B), B, 0, stream>>>(ei, hh, alpha, gat, E, Etot);

  // ---- global mean pool + MLP head ----
  gg_fill_f32<<<gg_nblk((long long)NGRAPH * GH, B), B, 0, stream>>>(pool, 0.f, (long long)NGRAPH * GH);
  gg_fill_f32<<<1, B, 0, stream>>>(cnt, 0.f, NGRAPH);
  gg_pool<<<gg_nblk((long long)N * GH, B), B, 0, stream>>>(gat, bg, batch, pool, cnt, N);
  gg_mlp<<<1, 64, 0, stream>>>(pool, cnt, Wf1, bf1, Wf2, bf2, out);
}
